// SelfAttn_34548716929501
// MI455X (gfx1250) — matrix operand order repaired and checked by reading, not compiled, not run on hardware
//
#include <hip/hip_runtime.h>

typedef __attribute__((ext_vector_type(16))) __bf16 v16bf;
typedef __attribute__((ext_vector_type(8)))  float  v8f;
typedef __attribute__((ext_vector_type(8)))  __bf16 bf16x8;
typedef __attribute__((ext_vector_type(4)))  __bf16 bf16x4;

#define F_DIM   512
#define NHEAD   4
#define DQDIM   128
#define BATCH   1024
#define NTOK    168
#define JN      21
#define VPAD    176   // 11 tiles of 16
#define KPAD    192   // 6 k-steps of 32
#define LSTRIDE 200   // LDS row stride (elements), keeps 16B alignment
#define NEGM    (-4294967295.0f)

union FragU { v16bf f; bf16x8 h[2]; };

static __device__ __forceinline__ v8f vzero8() {
    v8f z;
#pragma unroll
    for (int i = 0; i < 8; ++i) z[i] = 0.0f;
    return z;
}

static __device__ __forceinline__ v8f wmma_bf16(v16bf a, v16bf b, v8f c) {
    // (neg_a, A, neg_b, B, c_mod, C, reuse_a, reuse_b)
    return __builtin_amdgcn_wmma_f32_16x16x32_bf16(false, a, false, b, (short)0, c, false, false);
}

// ---------------------------------------------------------------------------
// float -> bf16 conversion (weights)
// ---------------------------------------------------------------------------
__global__ void cvt_kernel(const float* __restrict__ in, __bf16* __restrict__ out, int n) {
    for (int i = blockIdx.x * blockDim.x + threadIdx.x; i < n; i += gridDim.x * blockDim.x)
        out[i] = (__bf16)in[i];
}

// ---------------------------------------------------------------------------
// LayerNorm over 512-wide rows: one wave (32 lanes) per row, bf16 output
// ---------------------------------------------------------------------------
__global__ __launch_bounds__(256) void ln_kernel(const float* __restrict__ x,
                                                 const float* __restrict__ g,
                                                 const float* __restrict__ bta,
                                                 __bf16* __restrict__ out, int rows) {
    const int wave = threadIdx.x >> 5, lane = threadIdx.x & 31;
    const int row = blockIdx.x * 8 + wave;
    if (row >= rows) return;

    const float4* xr = (const float4*)(x + (size_t)row * F_DIM);
    float4 vv[4];
    float s = 0.f, s2 = 0.f;
#pragma unroll
    for (int j = 0; j < 4; ++j) {
        vv[j] = xr[j * 32 + lane];
        s  += vv[j].x + vv[j].y + vv[j].z + vv[j].w;
        s2 += vv[j].x * vv[j].x + vv[j].y * vv[j].y + vv[j].z * vv[j].z + vv[j].w * vv[j].w;
    }
#pragma unroll
    for (int m = 1; m < 32; m <<= 1) {
        s  += __shfl_xor(s, m, 32);
        s2 += __shfl_xor(s2, m, 32);
    }
    const float mu = s * (1.0f / F_DIM);
    const float var = s2 * (1.0f / F_DIM) - mu * mu;
    const float rs = rsqrtf(var + 1e-6f);

    const float4* g4 = (const float4*)g;
    const float4* b4 = (const float4*)bta;
    bf16x4* orow = (bf16x4*)(out + (size_t)row * F_DIM);
#pragma unroll
    for (int j = 0; j < 4; ++j) {
        float4 gg = g4[j * 32 + lane];
        float4 bb = b4[j * 32 + lane];
        bf16x4 r;
        r[0] = (__bf16)((vv[j].x - mu) * rs * gg.x + bb.x);
        r[1] = (__bf16)((vv[j].y - mu) * rs * gg.y + bb.y);
        r[2] = (__bf16)((vv[j].z - mu) * rs * gg.z + bb.z);
        r[3] = (__bf16)((vv[j].w - mu) * rs * gg.w + bb.w);
        orow[j * 32 + lane] = r;
    }
}

// ---------------------------------------------------------------------------
// GEMM: C[M,N] = A[M,K](bf16) * W[N,K]^T(bf16) + bias (+resid) (relu?)
// 256 threads = 8 waves; wave tile 32(M) x 128(N); block tile 64(M) x 512(N)
// (block covers full N=512: weights read once per block, L2-resident)
// Per k-step: 4 A loads + 16 B loads -> 16 WMMAs (A frag reused 8x, B 2x).
// ---------------------------------------------------------------------------
template <bool RELU, bool RESID, bool OUTBF>
__global__ __launch_bounds__(256) void gemm_kernel(const __bf16* __restrict__ A,
                                                   const __bf16* __restrict__ W,
                                                   const float* __restrict__ bias,
                                                   const float* resid,
                                                   float* outF, __bf16* outB,
                                                   int Mrows, int N, int K) {
    const int tid  = threadIdx.x;
    const int wave = tid >> 5, lane = tid & 31;
    const int half = lane >> 4, ln = lane & 15;
    const int waveM = wave & 1;    // 0..1
    const int waveN = wave >> 1;   // 0..3
    const int row0 = blockIdx.y * 64 + waveM * 32;
    const int col0 = waveN * 128;

    v8f acc[2][8];
#pragma unroll
    for (int m = 0; m < 2; ++m)
#pragma unroll
        for (int t = 0; t < 8; ++t) acc[m][t] = vzero8();

    const __bf16* arow0 = A + (size_t)(row0 + ln) * K;
    const __bf16* arow1 = A + (size_t)(row0 + 16 + ln) * K;
    for (int k0 = 0; k0 < K; k0 += 32) {
        FragU a0, a1;
        a0.h[0] = *(const bf16x8*)(arow0 + k0 + 8 * half);      // K = k0 + 8h + i      (i<8)
        a0.h[1] = *(const bf16x8*)(arow0 + k0 + 16 + 8 * half); // K = k0 + 16 + 8h + i (i>=8)
        a1.h[0] = *(const bf16x8*)(arow1 + k0 + 8 * half);
        a1.h[1] = *(const bf16x8*)(arow1 + k0 + 16 + 8 * half);
#pragma unroll
        for (int t = 0; t < 8; ++t) {
            const __bf16* wrow = W + (size_t)(col0 + t * 16 + ln) * K + k0;
            FragU b;
            b.h[0] = *(const bf16x8*)(wrow + 8 * half);      // K = k0 + 8h + i
            b.h[1] = *(const bf16x8*)(wrow + 16 + 8 * half);
            acc[0][t] = wmma_bf16(a0.f, b.f, acc[0][t]);
            acc[1][t] = wmma_bf16(a1.f, b.f, acc[1][t]);
        }
    }

#pragma unroll
    for (int t = 0; t < 8; ++t) {
        const int col = col0 + t * 16 + ln;
        const float bs = bias[col];
#pragma unroll
        for (int m = 0; m < 2; ++m) {
#pragma unroll
            for (int r = 0; r < 8; ++r) {
                const int row = row0 + m * 16 + r + 8 * half;
                float val = acc[m][t][r] + bs;
                if (RESID) val += resid[(size_t)row * N + col];
                if (RELU)  val = fmaxf(val, 0.0f);
                if (OUTBF) outB[(size_t)row * N + col] = (__bf16)val;
                else       outF[(size_t)row * N + col] = val;
            }
        }
    }
}

// ---------------------------------------------------------------------------
// Attention: one block per (dv-half, head, batch); 128 threads = 4 waves.
// Each wave processes query row-tiles rt = wave, wave+4, ...  (11 tiles).
// Scores = q k^T / sqrt(128) + mask; softmax; O = P V (WMMA bf16 throughout).
// ---------------------------------------------------------------------------
__global__ __launch_bounds__(128) void attn_kernel(const __bf16* __restrict__ q,
                                                   const __bf16* __restrict__ k,
                                                   const __bf16* __restrict__ v,
                                                   const int* __restrict__ valid,
                                                   __bf16* __restrict__ out) {
    __shared__ __align__(16) __bf16 vt[64][LSTRIDE];       // V^T for this dv-half
    __shared__ __align__(16) __bf16 pbuf[4][16][LSTRIDE];  // per-wave P staging
    __shared__ float maskv[KPAD];

    const int dvh  = blockIdx.x;   // 0..1 (which 64 of the 128 dv columns)
    const int head = blockIdx.y;
    const int b    = blockIdx.z;
    const int tid  = threadIdx.x;
    const size_t baseTok = (size_t)b * NTOK;
    const int colbase = head * DQDIM + dvh * 64;

    // stage transposed V (dv fastest -> coalesced global reads), zero padding
    for (int i = tid; i < 64 * KPAD; i += 128) {
        const int tok = i >> 6, dv = i & 63;
        __bf16 val = (__bf16)0.0f;
        if (tok < NTOK) val = v[(baseTok + tok) * F_DIM + colbase + dv];
        vt[dv][tok] = val;
    }
    // per-token additive mask (pad tokens get 2*NEG so they always vanish)
    for (int i = tid; i < KPAD; i += 128) {
        float w;
        if (i < NTOK) w = valid[b * 8 + i / JN] ? 0.0f : NEGM;
        else          w = 2.0f * NEGM;
        maskv[i] = w;
    }
    __syncthreads();

    const int wave = tid >> 5, lane = tid & 31;
    const int half = lane >> 4, ln = lane & 15;

    // zero P columns [NTOK, KPAD) once (own wave's buffer)
    for (int i = lane; i < 16 * (KPAD - NTOK); i += 32) {
        const int rr = i / (KPAD - NTOK), cc = NTOK + i % (KPAD - NTOK);
        pbuf[wave][rr][cc] = (__bf16)0.0f;
    }

    const float scale = 0.08838834764831845f;  // 1/sqrt(128)

    for (int rt = wave; rt < 11; rt += 4) {
        // q A-fragments for this 16-query tile (clamp padded rows)
        FragU qa[4];
        {
            int tok = rt * 16 + ln;
            if (tok > NTOK - 1) tok = NTOK - 1;
            const __bf16* qrow = q + (baseTok + tok) * F_DIM + head * DQDIM;
#pragma unroll
            for (int kq = 0; kq < 4; ++kq) {
                qa[kq].h[0] = *(const bf16x8*)(qrow + kq * 32 + 8 * half);
                qa[kq].h[1] = *(const bf16x8*)(qrow + kq * 32 + 16 + 8 * half);
            }
        }
        // scores: 11 column tiles, K=128 in 4 wmma steps
        v8f sc[11];
#pragma unroll
        for (int c = 0; c < 11; ++c) {
            v8f s = vzero8();
            int tokn = c * 16 + ln;
            if (tokn > NTOK - 1) tokn = NTOK - 1;
            const __bf16* krow = k + (baseTok + tokn) * F_DIM + head * DQDIM;
#pragma unroll
            for (int kq = 0; kq < 4; ++kq) {
                FragU kb;
                kb.h[0] = *(const bf16x8*)(krow + kq * 32 + 8 * half);
                kb.h[1] = *(const bf16x8*)(krow + kq * 32 + 16 + 8 * half);
                s = wmma_bf16(qa[kq].f, kb.f, s);
            }
            sc[c] = s;
        }
        // scale + additive mask
        float qm[8];
#pragma unroll
        for (int r = 0; r < 8; ++r) qm[r] = maskv[rt * 16 + r + 8 * half];
#pragma unroll
        for (int c = 0; c < 11; ++c) {
            const float km = maskv[c * 16 + ln];
#pragma unroll
            for (int r = 0; r < 8; ++r) sc[c][r] = sc[c][r] * scale + km + qm[r];
        }
        // softmax per row (row lives in 16 lanes of one half) -> P to LDS (bf16)
#pragma unroll
        for (int r = 0; r < 8; ++r) {
            float mx = -3.4e38f;
#pragma unroll
            for (int c = 0; c < 11; ++c) mx = fmaxf(mx, sc[c][r]);
#pragma unroll
            for (int m = 1; m < 16; m <<= 1) mx = fmaxf(mx, __shfl_xor(mx, m, 32));
            float e[11], sum = 0.0f;
#pragma unroll
            for (int c = 0; c < 11; ++c) { e[c] = __expf(sc[c][r] - mx); sum += e[c]; }
#pragma unroll
            for (int m = 1; m < 16; m <<= 1) sum += __shfl_xor(sum, m, 32);
            const float inv = 1.0f / sum;
            const int mrow = r + 8 * half;
#pragma unroll
            for (int c = 0; c < 11; ++c)
                pbuf[wave][mrow][c * 16 + ln] = (__bf16)(e[c] * inv);
        }
        // make cross-lane LDS writes visible within this wave
        __asm__ volatile("s_wait_dscnt 0" ::: "memory");

        // O = P x V : 4 n-tiles of 16 dv cols, K = 192 in 6 wmma steps
        v8f ob[4];
#pragma unroll
        for (int t = 0; t < 4; ++t) ob[t] = vzero8();
#pragma unroll
        for (int ks = 0; ks < 6; ++ks) {
            FragU pa;
            const __bf16* prow = &pbuf[wave][ln][ks * 32];
            pa.h[0] = *(const bf16x8*)(prow + 8 * half);
            pa.h[1] = *(const bf16x8*)(prow + 16 + 8 * half);
#pragma unroll
            for (int t = 0; t < 4; ++t) {
                FragU vb;
                const __bf16* vrow = &vt[t * 16 + ln][ks * 32];
                vb.h[0] = *(const bf16x8*)(vrow + 8 * half);
                vb.h[1] = *(const bf16x8*)(vrow + 16 + 8 * half);
                ob[t] = wmma_bf16(pa.f, vb.f, ob[t]);
            }
        }
        // store (skip padded query rows)
#pragma unroll
        for (int t = 0; t < 4; ++t) {
#pragma unroll
            for (int r = 0; r < 8; ++r) {
                const int tok = rt * 16 + r + 8 * half;
                if (tok < NTOK)
                    out[(baseTok + tok) * F_DIM + colbase + t * 16 + ln] = (__bf16)ob[t][r];
            }
        }
    }
}

// ---------------------------------------------------------------------------
extern "C" void kernel_launch(void* const* d_in, const int* in_sizes, int n_in,
                              void* d_out, int out_size, void* d_ws, size_t ws_size,
                              hipStream_t stream) {
    (void)in_sizes; (void)n_in; (void)out_size; (void)ws_size;

    const float* x     = (const float*)d_in[0];
    const int*   valid = (const int*)d_in[1];
    const float* wq    = (const float*)d_in[2];
    const float* bq    = (const float*)d_in[3];
    const float* wk    = (const float*)d_in[4];
    const float* bk    = (const float*)d_in[5];
    const float* wv    = (const float*)d_in[6];
    const float* bv    = (const float*)d_in[7];
    const float* fc_w  = (const float*)d_in[8];
    const float* fc_b  = (const float*)d_in[9];
    const float* ln1_g = (const float*)d_in[10];
    const float* ln1_b = (const float*)d_in[11];
    const float* ln2_g = (const float*)d_in[12];
    const float* ln2_b = (const float*)d_in[13];
    const float* fc1_w = (const float*)d_in[14];
    const float* fc1_b = (const float*)d_in[15];
    const float* fc2_w = (const float*)d_in[16];
    const float* fc2_b = (const float*)d_in[17];

    const int Mrows = BATCH * NTOK;                 // 172032, divisible by 64
    const size_t szb = (size_t)Mrows * F_DIM;       // elements per activation buffer
    const size_t wsz = (size_t)F_DIM * F_DIM;       // elements per weight

    char* p = (char*)d_ws;
    __bf16* hb   = (__bf16*)p; p += szb * 2;   // LN1 out; reused for LN2 out
    __bf16* qb   = (__bf16*)p; p += szb * 2;   // Q; reused for MLP hidden
    __bf16* kb   = (__bf16*)p; p += szb * 2;   // K
    __bf16* vb   = (__bf16*)p; p += szb * 2;   // V
    __bf16* ab   = (__bf16*)p; p += szb * 2;   // attention output
    __bf16* wqb  = (__bf16*)p; p += wsz * 2;
    __bf16* wkb  = (__bf16*)p; p += wsz * 2;
    __bf16* wvb  = (__bf16*)p; p += wsz * 2;
    __bf16* fcwb = (__bf16*)p; p += wsz * 2;
    __bf16* fc1b = (__bf16*)p; p += wsz * 2;
    __bf16* fc2b = (__bf16*)p; p += wsz * 2;
    float* xout = (float*)d_out;                // holds x2 then final result

    // weights -> bf16 (small, L2-resident thereafter)
    cvt_kernel<<<256, 256, 0, stream>>>(wq,    wqb,  (int)wsz);
    cvt_kernel<<<256, 256, 0, stream>>>(wk,    wkb,  (int)wsz);
    cvt_kernel<<<256, 256, 0, stream>>>(wv,    wvb,  (int)wsz);
    cvt_kernel<<<256, 256, 0, stream>>>(fc_w,  fcwb, (int)wsz);
    cvt_kernel<<<256, 256, 0, stream>>>(fc1_w, fc1b, (int)wsz);
    cvt_kernel<<<256, 256, 0, stream>>>(fc2_w, fc2b, (int)wsz);

    // LN1
    ln_kernel<<<Mrows / 8, 256, 0, stream>>>(x, ln1_g, ln1_b, hb, Mrows);

    const dim3 gg(1, Mrows / 64);
    // QKV projections
    gemm_kernel<false, false, true><<<gg, 256, 0, stream>>>(hb, wqb, bq, nullptr, nullptr, qb, Mrows, F_DIM, F_DIM);
    gemm_kernel<false, false, true><<<gg, 256, 0, stream>>>(hb, wkb, bk, nullptr, nullptr, kb, Mrows, F_DIM, F_DIM);
    gemm_kernel<false, false, true><<<gg, 256, 0, stream>>>(hb, wvb, bv, nullptr, nullptr, vb, Mrows, F_DIM, F_DIM);

    // attention
    attn_kernel<<<dim3(2, NHEAD, BATCH), 128, 0, stream>>>(qb, kb, vb, valid, ab);

    // output projection + residual -> x2 (f32, staged in d_out)
    gemm_kernel<false, true, false><<<gg, 256, 0, stream>>>(ab, fcwb, fc_b, x, xout, nullptr, Mrows, F_DIM, F_DIM);

    // LN2 (reads x2 from d_out)
    ln_kernel<<<Mrows / 8, 256, 0, stream>>>(xout, ln2_g, ln2_b, hb, Mrows);

    // MLP: fc1 + relu (bf16), fc2 + residual (in-place on d_out)
    gemm_kernel<true,  false, true ><<<gg, 256, 0, stream>>>(hb, fc1b, fc1_b, nullptr, nullptr, qb, Mrows, F_DIM, F_DIM);
    gemm_kernel<false, true,  false><<<gg, 256, 0, stream>>>(qb, fc2b, fc2_b, xout, xout, nullptr, Mrows, F_DIM, F_DIM);
}
